// GNNEncoder_50852412784898
// MI455X (gfx1250) — matrix operand order repaired and checked by reading, not compiled, not run on hardware
//
#include <hip/hip_runtime.h>
#include <hip/hip_bf16.h>

#define NN  50000
#define EE  800000
#define DD  64
#define FFD 256
#define GG  512
#define NTL 3125   // NN/16 row tiles

typedef __attribute__((ext_vector_type(16))) __bf16 v16bf;
typedef __attribute__((ext_vector_type(8)))  __bf16 v8bf;
typedef __attribute__((ext_vector_type(8)))  float  v8f;

// ---------------------------------------------------------------- WMMA helpers
__device__ __forceinline__ v8f wmma_bf16(v16bf a, v16bf b, v8f c) {
  // (neg_a, A, neg_b, B, c_mod, C, reuse_a, reuse_b)
  return __builtin_amdgcn_wmma_f32_16x16x32_bf16(false, a, false, b, (short)0, c,
                                                 false, false);
}

// A-matrix 16x32 bf16 fragment (ISA 7.12.2): lane l holds row (l&15);
// element e -> K = (e%8) + (e>=8 ? 16 : 0) + (lane>=16 ? 8 : 0)
__device__ __forceinline__ v16bf load_a_f32(const float* __restrict__ row, int kt, int hi) {
  int base = 32 * kt + (hi ? 8 : 0);
  const float4 a0 = *reinterpret_cast<const float4*>(row + base);
  const float4 a1 = *reinterpret_cast<const float4*>(row + base + 4);
  const float4 b0 = *reinterpret_cast<const float4*>(row + base + 16);
  const float4 b1 = *reinterpret_cast<const float4*>(row + base + 20);
  v16bf a;
  a[0] = (__bf16)a0.x;  a[1] = (__bf16)a0.y;  a[2]  = (__bf16)a0.z;  a[3]  = (__bf16)a0.w;
  a[4] = (__bf16)a1.x;  a[5] = (__bf16)a1.y;  a[6]  = (__bf16)a1.z;  a[7]  = (__bf16)a1.w;
  a[8] = (__bf16)b0.x;  a[9] = (__bf16)b0.y;  a[10] = (__bf16)b0.z;  a[11] = (__bf16)b0.w;
  a[12] = (__bf16)b1.x; a[13] = (__bf16)b1.y; a[14] = (__bf16)b1.z; a[15] = (__bf16)b1.w;
  return a;
}

// A fragment from bf16 LDS staging (row-major, 16B-aligned vector loads -> ds_load_b128)
__device__ __forceinline__ v16bf load_a_lds(const __bf16* row, int kt, int hi) {
  int base = 32 * kt + (hi ? 8 : 0);
  v8bf lo = *reinterpret_cast<const v8bf*>(row + base);
  v8bf hh = *reinterpret_cast<const v8bf*>(row + base + 16);
  return __builtin_shufflevector(lo, hh, 0, 1, 2, 3, 4, 5, 6, 7,
                                 8, 9, 10, 11, 12, 13, 14, 15);
}

// B fragment from pre-swizzled weight buffer: one contiguous 32B vector per lane.
__device__ __forceinline__ v16bf load_b_frag(const __bf16* __restrict__ wf, int frag, int lane) {
  return *reinterpret_cast<const v16bf*>(wf + ((size_t)frag * 32 + lane) * 16);
}

__device__ __forceinline__ v8f splat8(float v) {
  v8f r;
#pragma unroll
  for (int i = 0; i < 8; i++) r[i] = v;
  return r;
}

// LayerNorm over a 16x64 tile in C-layout: acc[n][r] is (row r+hi*8, col n*16+ln15).
__device__ __forceinline__ void ln_rows(v8f acc[4], const float* __restrict__ gam,
                                        const float* __restrict__ bet, int ln15) {
#pragma unroll
  for (int r = 0; r < 8; r++) {
    float s = acc[0][r] + acc[1][r] + acc[2][r] + acc[3][r];
#pragma unroll
    for (int off = 1; off < 16; off <<= 1) s += __shfl_xor(s, off, 32);
    float mean = s * (1.0f / 64.0f);
    float vs = 0.0f;
#pragma unroll
    for (int n = 0; n < 4; n++) { float d = acc[n][r] - mean; vs += d * d; }
#pragma unroll
    for (int off = 1; off < 16; off <<= 1) vs += __shfl_xor(vs, off, 32);
    float inv = rsqrtf(vs * (1.0f / 64.0f) + 1e-5f);
#pragma unroll
    for (int n = 0; n < 4; n++) {
      int col = n * 16 + ln15;
      acc[n][r] = (acc[n][r] - mean) * inv * gam[col] + bet[col];
    }
  }
}

// ---------------------------------------------------------------- weight swizzle
// Pack W[K][Ncols] (f32, row-major) into bf16 WMMA-B fragments:
// out[((n*KT + kt)*32 + lane)*16 + e] = W[32*kt + (lane>=16?8:0) + e + (e>=8?8:0)][n*16 + (lane&15)]
__global__ void k_swizzle(const float* __restrict__ w, __bf16* __restrict__ out,
                          int K, int Ncols) {
  int idx = blockIdx.x * blockDim.x + threadIdx.x;
  if (idx >= K * Ncols) return;
  int KT = K >> 5;
  int e = idx & 15;
  int lane = (idx >> 4) & 31;
  int t = idx >> 9;           // n*KT + kt
  int kt = t % KT;
  int n = t / KT;
  int col = n * 16 + (lane & 15);
  int krow = 32 * kt + ((lane >> 4) ? 8 : 0) + e + (e >= 8 ? 8 : 0);
  out[idx] = (__bf16)w[(size_t)krow * Ncols + col];
}

// ---------------------------------------------------------------- kernels
__global__ void k_encode(const float* __restrict__ x, const float* __restrict__ ew,
                         const float* __restrict__ eb, float* __restrict__ h,
                         float* __restrict__ agg) {
  int idx = blockIdx.x * blockDim.x + threadIdx.x;
  if (idx >= NN * DD) return;
  int n = idx >> 6, d = idx & 63;
  float v = fmaxf(x[n] * ew[d] + eb[d], 0.0f);
  h[idx] = v;
  agg[idx] = v;
}

__global__ void k_scatter(const int* __restrict__ ei, const float* __restrict__ h,
                          float* __restrict__ agg) {
  int idx = blockIdx.x * blockDim.x + threadIdx.x;
  if (idx >= EE * 16) return;
  int e = idx >> 4, c4 = (idx & 15) * 4;
  int s = ei[e];
  int d = ei[EE + e];
  const float4 v = *reinterpret_cast<const float4*>(h + (size_t)s * DD + c4);
  float* p = agg + (size_t)d * DD + c4;
  atomicAdd(p + 0, v.x);
  atomicAdd(p + 1, v.y);
  atomicAdd(p + 2, v.z);
  atomicAdd(p + 3, v.w);
}

// One wave32 per 16-row tile: GIN MLP + LN1 + FF MLP + LN2, fully fused.
__global__ __launch_bounds__(128) void k_block(
    const float* __restrict__ agg, const float* __restrict__ hin,
    const __bf16* __restrict__ w1f, const float* __restrict__ b1,
    const __bf16* __restrict__ w2f, const float* __restrict__ b2,
    const float* __restrict__ g1, const float* __restrict__ be1,
    const __bf16* __restrict__ fw1f, const float* __restrict__ fb1,
    const __bf16* __restrict__ fw2f, const float* __restrict__ fb2,
    const float* __restrict__ g2, const float* __restrict__ be2,
    float* __restrict__ hout, float* __restrict__ aggout) {
  __shared__ __align__(32) __bf16 stage[4][16 * FFD];  // 8KB per wave staging
  int wave = threadIdx.x >> 5;
  int lane = threadIdx.x & 31;
  int tile = blockIdx.x * 4 + wave;
  if (tile >= NTL) return;  // wave-uniform -> EXEC stays all-ones for WMMA
  int ln15 = lane & 15, hi = lane >> 4;
  int row0 = tile * 16;
  __bf16* st = &stage[wave][0];

  // ---- GIN GEMM1: t1 = relu(agg @ w1 + b1), staged to LDS (16x64 bf16) ----
  const float* arow = agg + (size_t)(row0 + ln15) * DD;
  v16bf A0 = load_a_f32(arow, 0, hi);
  v16bf A1 = load_a_f32(arow, 1, hi);
#pragma unroll
  for (int n = 0; n < 4; n++) {
    int col = n * 16 + ln15;
    v8f acc = splat8(b1[col]);
    acc = wmma_bf16(A0, load_b_frag(w1f, n * 2 + 0, lane), acc);
    acc = wmma_bf16(A1, load_b_frag(w1f, n * 2 + 1, lane), acc);
#pragma unroll
    for (int r = 0; r < 8; r++)
      st[(r + hi * 8) * DD + col] = (__bf16)fmaxf(acc[r], 0.0f);
  }

  // ---- GIN GEMM2: g = t1 @ w2 + b2 + h ; hmid = LN1(g) ----
  v16bf T0 = load_a_lds(st + ln15 * DD, 0, hi);
  v16bf T1 = load_a_lds(st + ln15 * DD, 1, hi);
  v8f hmid[4];
#pragma unroll
  for (int n = 0; n < 4; n++) {
    int col = n * 16 + ln15;
    v8f acc = splat8(b2[col]);
    acc = wmma_bf16(T0, load_b_frag(w2f, n * 2 + 0, lane), acc);
    acc = wmma_bf16(T1, load_b_frag(w2f, n * 2 + 1, lane), acc);
#pragma unroll
    for (int r = 0; r < 8; r++)
      acc[r] += hin[(size_t)(row0 + r + hi * 8) * DD + col];  // residual g + h
    hmid[n] = acc;
  }
  ln_rows(hmid, g1, be1, ln15);

  // stage hmid as bf16 16x64 for FF GEMM1 A-fragments
#pragma unroll
  for (int n = 0; n < 4; n++) {
    int col = n * 16 + ln15;
#pragma unroll
    for (int r = 0; r < 8; r++) st[(r + hi * 8) * DD + col] = (__bf16)hmid[n][r];
  }
  v16bf H0 = load_a_lds(st + ln15 * DD, 0, hi);
  v16bf H1 = load_a_lds(st + ln15 * DD, 1, hi);

  // ---- FF GEMM1: t2 = relu(hmid @ fw1 + fb1), staged 16x256 bf16 ----
  for (int n = 0; n < 16; n++) {
    int col = n * 16 + ln15;
    v8f acc = splat8(fb1[col]);
    acc = wmma_bf16(H0, load_b_frag(fw1f, n * 2 + 0, lane), acc);
    acc = wmma_bf16(H1, load_b_frag(fw1f, n * 2 + 1, lane), acc);
#pragma unroll
    for (int r = 0; r < 8; r++)
      st[(r + hi * 8) * FFD + col] = (__bf16)fmaxf(acc[r], 0.0f);
  }

  // ---- FF GEMM2: out = LN2(t2 @ fw2 + fb2 + hmid) ----
  v8f outv[4];
#pragma unroll
  for (int n = 0; n < 4; n++) outv[n] = splat8(fb2[n * 16 + ln15]);
#pragma unroll
  for (int kt = 0; kt < 8; kt++) {
    v16bf A = load_a_lds(st + ln15 * FFD, kt, hi);
#pragma unroll
    for (int n = 0; n < 4; n++)
      outv[n] = wmma_bf16(A, load_b_frag(fw2f, n * 8 + kt, lane), outv[n]);
  }
#pragma unroll
  for (int n = 0; n < 4; n++) {
#pragma unroll
    for (int r = 0; r < 8; r++) outv[n][r] += hmid[n][r];
  }
  ln_rows(outv, g2, be2, ln15);

#pragma unroll
  for (int n = 0; n < 4; n++) {
    int col = n * 16 + ln15;
#pragma unroll
    for (int r = 0; r < 8; r++) {
      size_t o = (size_t)(row0 + r + hi * 8) * DD + col;
      float v = outv[n][r];
      hout[o] = v;
      aggout[o] = v;  // pre-init next layer's aggregation buffer to h
    }
  }
}

// ---------------------------------------------------------------- pooling
__device__ __forceinline__ unsigned fenc(float f) {
  unsigned u = __float_as_uint(f);
  return (u & 0x80000000u) ? ~u : (u | 0x80000000u);
}
__device__ __forceinline__ float fdec(unsigned e) {
  unsigned u = (e & 0x80000000u) ? (e & 0x7fffffffu) : ~e;
  return __uint_as_float(u);
}

__global__ void k_pool_init(float* sums, unsigned* maxb, float* counts) {
  int idx = blockIdx.x * blockDim.x + threadIdx.x;
  if (idx < GG * DD) {
    sums[idx] = 0.0f;
    maxb[idx] = 0x007fffffu;  // enc(-inf): empty segments decode to -inf
  }
  if (idx < GG) counts[idx] = 0.0f;
}

__global__ void k_pool_accum(const float* __restrict__ h, const int* __restrict__ batch,
                             float* sums, unsigned* maxb, float* counts) {
  int idx = blockIdx.x * blockDim.x + threadIdx.x;
  if (idx >= NN * DD) return;
  int n = idx >> 6, d = idx & 63;
  int b = batch[n];
  float v = h[idx];
  atomicAdd(&sums[b * DD + d], v);
  atomicMax(&maxb[b * DD + d], fenc(v));
  if (d == 0) atomicAdd(&counts[b], 1.0f);
}

__global__ void k_pool_final(const float* __restrict__ sums, const unsigned* __restrict__ maxb,
                             const float* __restrict__ counts, float* __restrict__ out) {
  int idx = blockIdx.x * blockDim.x + threadIdx.x;
  if (idx >= GG * DD) return;
  int g = idx >> 6, d = idx & 63;
  out[g * 2 * DD + d] = sums[idx] / fmaxf(counts[g], 1.0f);
  out[g * 2 * DD + DD + d] = fdec(maxb[idx]);
}

__global__ void k_copy(const float* __restrict__ h, float* __restrict__ out) {
  int idx = blockIdx.x * blockDim.x + threadIdx.x;
  if (idx >= NN * DD) return;
  out[idx] = h[idx];
}

// ---------------------------------------------------------------- launch
extern "C" void kernel_launch(void* const* d_in, const int* in_sizes, int n_in,
                              void* d_out, int out_size, void* d_ws, size_t ws_size,
                              hipStream_t stream) {
  const float* x     = (const float*)d_in[0];
  const int*   ei    = (const int*)d_in[1];
  const int*   batch = (const int*)d_in[2];
  const float* ew    = (const float*)d_in[3];
  const float* eb    = (const float*)d_in[4];
  const float* gw1   = (const float*)d_in[5];
  const float* gb1   = (const float*)d_in[6];
  const float* gw2   = (const float*)d_in[7];
  const float* gb2   = (const float*)d_in[8];
  const float* l1g   = (const float*)d_in[9];
  const float* l1b   = (const float*)d_in[10];
  const float* fw1   = (const float*)d_in[11];
  const float* fb1   = (const float*)d_in[12];
  const float* fw2   = (const float*)d_in[13];
  const float* fb2   = (const float*)d_in[14];
  const float* l2g   = (const float*)d_in[15];
  const float* l2b   = (const float*)d_in[16];
  float* out = (float*)d_out;

  // workspace layout (base is 256B aligned; all sections keep 32B alignment)
  float*    h      = (float*)d_ws;
  float*    agg    = h + (size_t)NN * DD;
  float*    sums   = agg + (size_t)NN * DD;
  float*    counts = sums + GG * DD;        // GG*DD = 32768 floats
  unsigned* maxb   = (unsigned*)(counts + GG);
  __bf16*   w1f    = (__bf16*)(maxb + GG * DD);          // 3 * 4096 bf16
  __bf16*   w2f    = w1f + 3 * (size_t)DD * DD;          // 3 * 4096 bf16
  __bf16*   fw1f   = w2f + 3 * (size_t)DD * DD;          // 3 * 16384 bf16
  __bf16*   fw2f   = fw1f + 3 * (size_t)DD * FFD;        // 3 * 16384 bf16

  const int TB = 256;

  // pre-swizzle all weight matrices into WMMA-B fragment layout (bf16)
  for (int i = 0; i < 3; i++) {
    k_swizzle<<<(DD * DD + TB - 1) / TB, TB, 0, stream>>>(
        gw1 + (size_t)i * DD * DD, w1f + (size_t)i * DD * DD, DD, DD);
    k_swizzle<<<(DD * DD + TB - 1) / TB, TB, 0, stream>>>(
        gw2 + (size_t)i * DD * DD, w2f + (size_t)i * DD * DD, DD, DD);
    k_swizzle<<<(DD * FFD + TB - 1) / TB, TB, 0, stream>>>(
        fw1 + (size_t)i * DD * FFD, fw1f + (size_t)i * DD * FFD, DD, FFD);
    k_swizzle<<<(FFD * DD + TB - 1) / TB, TB, 0, stream>>>(
        fw2 + (size_t)i * FFD * DD, fw2f + (size_t)i * FFD * DD, FFD, DD);
  }

  k_encode<<<(NN * DD + TB - 1) / TB, TB, 0, stream>>>(x, ew, eb, h, agg);
  for (int i = 0; i < 3; i++) {
    k_scatter<<<(EE * 16 + TB - 1) / TB, TB, 0, stream>>>(ei, h, agg);
    k_block<<<(NTL + 3) / 4, 128, 0, stream>>>(
        agg, h,
        w1f + (size_t)i * DD * DD, gb1 + i * DD,
        w2f + (size_t)i * DD * DD, gb2 + i * DD,
        l1g + i * DD, l1b + i * DD,
        fw1f + (size_t)i * DD * FFD, fb1 + i * FFD,
        fw2f + (size_t)i * FFD * DD, fb2 + i * DD,
        l2g + i * DD, l2b + i * DD,
        h, agg);
  }
  k_pool_init<<<(GG * DD + TB - 1) / TB, TB, 0, stream>>>(sums, maxb, counts);
  k_pool_accum<<<(NN * DD + TB - 1) / TB, TB, 0, stream>>>(h, batch, sums, maxb, counts);
  k_pool_final<<<(GG * DD + TB - 1) / TB, TB, 0, stream>>>(sums, maxb, counts, out);
  k_copy<<<(NN * DD + TB - 1) / TB, TB, 0, stream>>>(h, out + (size_t)GG * 2 * DD);
}
